// GPTLanguageModel_45105746543106
// MI455X (gfx1250) — matrix operand order repaired and checked by reading, not compiled, not run on hardware
//
#include <hip/hip_runtime.h>
#include <hip/hip_bf16.h>
#include <stdint.h>

typedef __attribute__((ext_vector_type(16))) _Float16 v16h;
typedef __attribute__((ext_vector_type(8)))  _Float16 v8h;
typedef __attribute__((ext_vector_type(8)))  float    v8f;
typedef __attribute__((ext_vector_type(4)))  unsigned int v4u;
typedef __attribute__((ext_vector_type(8)))  int      v8i;
typedef __attribute__((ext_vector_type(4)))  int      v4i;

#define Dm   1024
#define Tn   1024
#define Bn   2
#define Mrows (Bn * Tn)   // 2048
#define Hn   16
#define HD   64
#define Fn   4096
#define Vn   32000
#define Ln   8

static __device__ inline v8f wmma16(v16h a, v16h b, v8f c) {
  return __builtin_amdgcn_wmma_f32_16x16x32_f16(
      false, a, false, b, (short)0, c, false, false);
}

static __device__ inline v16h cat8(v8h a, v8h b) {
  return __builtin_shufflevector(a, b, 0,1,2,3,4,5,6,7,8,9,10,11,12,13,14,15);
}

// ---------------------------------------------------------------------------
// f32 -> f16 conversion (weights, once per launch). n multiple of 4.
// ---------------------------------------------------------------------------
__global__ __launch_bounds__(256)
void cvt_kernel(const float* __restrict__ src, _Float16* __restrict__ dst) {
  size_t i4 = (size_t)blockIdx.x * 256 + threadIdx.x;
  float4 v = ((const float4*)src)[i4];
  _Float16* d = dst + i4 * 4;
  d[0] = (_Float16)v.x; d[1] = (_Float16)v.y;
  d[2] = (_Float16)v.z; d[3] = (_Float16)v.w;
}

// ---------------------------------------------------------------------------
// Embedding: x[b,t,:] = tok_emb[idx[b,t],:] + pos_emb[t,:]   (f32 residual)
// ---------------------------------------------------------------------------
__global__ __launch_bounds__(256)
void embed_kernel(const int* __restrict__ idx, const float* __restrict__ tok,
                  const float* __restrict__ pos, float* __restrict__ x) {
  int row = blockIdx.x;
  int t   = row & (Tn - 1);
  int tokid = idx[row];
  const float4* tp = (const float4*)(tok + (size_t)tokid * Dm);
  const float4* pp = (const float4*)(pos + (size_t)t * Dm);
  float4* xp = (float4*)(x + (size_t)row * Dm);
  int i = threadIdx.x;
  float4 a = tp[i], b = pp[i];
  float4 r; r.x = a.x + b.x; r.y = a.y + b.y; r.z = a.z + b.z; r.w = a.w + b.w;
  xp[i] = r;
}

// ---------------------------------------------------------------------------
// LayerNorm (D=1024): f32 in, f16 out (feeds WMMA GEMMs)
// ---------------------------------------------------------------------------
__global__ __launch_bounds__(256)
void layernorm_kernel(const float* __restrict__ x, const float* __restrict__ g,
                      const float* __restrict__ b, _Float16* __restrict__ out) {
  __shared__ float s_sum[256];
  __shared__ float s_sq[256];
  int row = blockIdx.x;
  int tid = threadIdx.x;
  const float4* xp = (const float4*)(x + (size_t)row * Dm);
  float4 v = xp[tid];
  s_sum[tid] = v.x + v.y + v.z + v.w;
  s_sq[tid]  = v.x*v.x + v.y*v.y + v.z*v.z + v.w*v.w;
  __syncthreads();
  for (int off = 128; off > 0; off >>= 1) {
    if (tid < off) { s_sum[tid] += s_sum[tid+off]; s_sq[tid] += s_sq[tid+off]; }
    __syncthreads();
  }
  float mean = s_sum[0] * (1.0f / Dm);
  float var  = s_sq[0] * (1.0f / Dm) - mean * mean;
  float rs   = rsqrtf(var + 1e-5f);
  float4 gg = ((const float4*)g)[tid];
  float4 bb = ((const float4*)b)[tid];
  _Float16* d = out + (size_t)row * Dm + tid * 4;
  d[0] = (_Float16)((v.x - mean) * rs * gg.x + bb.x);
  d[1] = (_Float16)((v.y - mean) * rs * gg.y + bb.y);
  d[2] = (_Float16)((v.z - mean) * rs * gg.z + bb.z);
  d[3] = (_Float16)((v.w - mean) * rs * gg.w + bb.w);
}

// ---------------------------------------------------------------------------
// WMMA GEMM: C[M,N] = act(A[M,K] @ W[K,N] + bias) (+ residual)
// A,W are f16. A tile staged via per-lane async global->LDS (ASYNCcnt);
// W tile staged via TDM tensor_load_to_lds with hardware LDS padding
// (TENSORcnt). 8 waves, 128x128 tile, 32x64 per wave.
// ---------------------------------------------------------------------------
template <bool HAS_BIAS, bool RELU, bool HAS_RES, bool OUT16>
__global__ __launch_bounds__(256)
void gemm_kernel(const _Float16* __restrict__ A, const _Float16* __restrict__ W,
                 const float* __restrict__ bias, const float* __restrict__ residual,
                 void* __restrict__ Cout, int N, int K) {
  __shared__ _Float16 Als[128][40];    // 128 x 32 (+8 pad) : 80B row stride
  __shared__ _Float16 Bls[32][136];    // 32 x 128 (+8 pad) : 272B row stride

  const int tid  = threadIdx.x;
  const int lane = tid & 31;
  const int wave = tid >> 5;
  const int wm   = wave >> 1;
  const int wn   = wave & 1;
  const int m0   = blockIdx.y * 128;
  const int n0   = blockIdx.x * 128;
  const int hi   = lane >> 4;
  const int lo   = lane & 15;

  v8f acc[2][4];
#pragma unroll
  for (int i = 0; i < 2; ++i)
#pragma unroll
    for (int j = 0; j < 4; ++j)
#pragma unroll
      for (int r = 0; r < 8; ++r) acc[i][j][r] = 0.0f;

  const int arow = tid >> 1, acs = (tid & 1) * 16;   // A: 2 threads per row

  // TDM descriptor for the B tile (2D, f16, 32 x 128, padded rows in LDS).
  // Built once; global_addr patched per K step.
  const uint32_t ldsB = (uint32_t)(uintptr_t)&Bls[0][0];
  v8i g1;
  {
    uint32_t dim0 = (uint32_t)N, dim1 = (uint32_t)K;
    g1[0] = (int)((1u << 16)      // data_size = 2 bytes
                | (1u << 20)      // pad_enable
                | (5u << 22)      // pad_interval: 64 DWORDs (256B row)
                | (3u << 25));    // pad_amount:   4 DWORDs (16B pad)
    g1[1] = (int)((dim0 & 0xFFFFu) << 16);               // tensor_dim0 lo
    g1[2] = (int)((dim0 >> 16) | ((dim1 & 0xFFFFu) << 16));
    g1[3] = (int)((dim1 >> 16) | (128u << 16));          // tile_dim0 = 128
    g1[4] = (int)32u;                                    // tile_dim1 = 32
    g1[5] = (int)dim0;                                   // tensor_dim0_stride
    g1[6] = 0; g1[7] = 0;
  }
  const v4i gz4 = {0, 0, 0, 0};
  const v8i gz8 = {0, 0, 0, 0, 0, 0, 0, 0};

  for (int k0 = 0; k0 < K; k0 += 32) {
    // ---- B tile via Tensor Data Mover (wave 0 issues one DMA) ----
    if (tid < 32) {
      uint64_t ga = (uint64_t)(uintptr_t)(W + (size_t)k0 * N + n0);
      v4u g0;
      g0[0] = 1u;                                        // count=1 (valid)
      g0[1] = ldsB;
      g0[2] = (uint32_t)ga;
      g0[3] = (uint32_t)((ga >> 32) & 0x01FFFFFFu) | (2u << 30);  // type=2
      __builtin_amdgcn_tensor_load_to_lds(g0, g1, gz4, gz4, gz8, 0);
    }
    // ---- A tile via async global->LDS (each thread: 32B = 2 x B128) ----
    {
      uint64_t ga = (uint64_t)(uintptr_t)(A + (size_t)(m0 + arow) * K + k0 + acs);
      uint32_t la = (uint32_t)(uintptr_t)&Als[arow][acs];
      asm volatile("global_load_async_to_lds_b128 %0, %1, off"
                   :: "v"(la), "v"(ga) : "memory");
      asm volatile("global_load_async_to_lds_b128 %0, %1, off offset:16"
                   :: "v"(la), "v"(ga) : "memory");
    }
    asm volatile("s_wait_asynccnt 0x0" ::: "memory");
    if (tid < 32) __builtin_amdgcn_s_wait_tensorcnt(0);
    __syncthreads();

    // ---- fragments ----
    v16h afrag[2];
#pragma unroll
    for (int s = 0; s < 2; ++s) {
      int r = wm * 32 + s * 16 + lo;
      const v8h* p = (const v8h*)&Als[r][hi * 16];       // K = hi*16 .. +7
      const v8h* q = (const v8h*)&Als[r][hi * 16 + 8];   // K = hi*16+8 .. +15
      afrag[s] = cat8(*p, *q);
    }
    v16h bfrag[4];
#pragma unroll
    for (int s = 0; s < 4; ++s) {
      int cidx = wn * 64 + s * 16 + lo;
      int kb = hi * 16;
#pragma unroll
      for (int i = 0; i < 16; ++i) bfrag[s][i] = Bls[kb + i][cidx];
    }
#pragma unroll
    for (int sm = 0; sm < 2; ++sm)
#pragma unroll
      for (int sn = 0; sn < 4; ++sn)
        acc[sm][sn] = wmma16(afrag[sm], bfrag[sn], acc[sm][sn]);

    __syncthreads();
  }

  // ---- epilogue ----
#pragma unroll
  for (int sm = 0; sm < 2; ++sm) {
#pragma unroll
    for (int sn = 0; sn < 4; ++sn) {
      int colg = n0 + wn * 64 + sn * 16 + lo;
      float bv = HAS_BIAS ? bias[colg] : 0.0f;
#pragma unroll
      for (int r = 0; r < 8; ++r) {
        int rowg = m0 + wm * 32 + sm * 16 + hi * 8 + r;
        float v = acc[sm][sn][r] + bv;
        if (RELU) v = fmaxf(v, 0.0f);
        if (HAS_RES) v += residual[(size_t)rowg * N + colg];
        if (OUT16) ((_Float16*)Cout)[(size_t)rowg * N + colg] = (_Float16)v;
        else       ((float*)Cout)   [(size_t)rowg * N + colg] = v;
      }
    }
  }
}

// ---------------------------------------------------------------------------
// Flash attention: one wave per (b, h, 16-row q tile). hd=64, causal.
// q,k,v,o are f16 [B*T, D]. Online softmax; P.V via zero-padded K=32 WMMA.
// ---------------------------------------------------------------------------
__global__ __launch_bounds__(32)
void attn_kernel(const _Float16* __restrict__ q, const _Float16* __restrict__ k,
                 const _Float16* __restrict__ v, _Float16* __restrict__ o) {
  __shared__ _Float16 pls[16][24];

  const int qt   = blockIdx.x;
  const int h    = blockIdx.y;
  const int bb   = blockIdx.z;
  const int lane = threadIdx.x;
  const int hi   = lane >> 4;
  const int lo   = lane & 15;
  const float scale = 0.125f;

  v16h qf[2];
#pragma unroll
  for (int kc = 0; kc < 2; ++kc) {
    size_t base = ((size_t)(bb * Tn + qt * 16 + lo)) * Dm + h * HD + kc * 32 + hi * 16;
    qf[kc] = cat8(*(const v8h*)(q + base), *(const v8h*)(q + base + 8));
  }

  v8f oacc[4];
#pragma unroll
  for (int n = 0; n < 4; ++n)
#pragma unroll
    for (int r = 0; r < 8; ++r) oacc[n][r] = 0.0f;
  float mrow[8], lrow[8];
#pragma unroll
  for (int r = 0; r < 8; ++r) { mrow[r] = -1e30f; lrow[r] = 0.0f; }

  for (int kt = 0; kt <= qt; ++kt) {
    v8f s;
#pragma unroll
    for (int r = 0; r < 8; ++r) s[r] = 0.0f;
#pragma unroll
    for (int kc = 0; kc < 2; ++kc) {
      size_t kbase = ((size_t)(bb * Tn + kt * 16 + lo)) * Dm + h * HD + kc * 32 + hi * 16;
      v16h bf = cat8(*(const v8h*)(k + kbase), *(const v8h*)(k + kbase + 8));
      s = wmma16(qf[kc], bf, s);
    }

    int key = kt * 16 + lo;
#pragma unroll
    for (int r = 0; r < 8; ++r) {
      int qrow = qt * 16 + hi * 8 + r;
      float sv = s[r] * scale;
      if (key > qrow) sv = -1e30f;
      float mx = sv;
      mx = fmaxf(mx, __shfl_xor(mx, 1, 32));
      mx = fmaxf(mx, __shfl_xor(mx, 2, 32));
      mx = fmaxf(mx, __shfl_xor(mx, 4, 32));
      mx = fmaxf(mx, __shfl_xor(mx, 8, 32));
      float mnew = fmaxf(mrow[r], mx);
      float corr = __expf(mrow[r] - mnew);
      float p    = __expf(sv - mnew);
      float ps   = p;
      ps += __shfl_xor(ps, 1, 32);
      ps += __shfl_xor(ps, 2, 32);
      ps += __shfl_xor(ps, 4, 32);
      ps += __shfl_xor(ps, 8, 32);
      lrow[r] = lrow[r] * corr + ps;
      mrow[r] = mnew;
#pragma unroll
      for (int n = 0; n < 4; ++n) oacc[n][r] *= corr;
      pls[hi * 8 + r][lo] = (_Float16)p;
    }

    v8h zero8;
#pragma unroll
    for (int i = 0; i < 8; ++i) zero8[i] = (_Float16)0.0f;
    v16h af = hi ? cat8(zero8, zero8) : cat8(*(const v8h*)&pls[lo][0], *(const v8h*)&pls[lo][8]);  // keys 16..31 = 0
#pragma unroll
    for (int n = 0; n < 4; ++n) {
      v16h vf;
      if (lane < 16) {
        size_t vbase = ((size_t)(bb * Tn + kt * 16)) * Dm + h * HD + n * 16 + lo;
#pragma unroll
        for (int i = 0; i < 16; ++i) vf[i] = v[vbase + (size_t)i * Dm];
      } else {
#pragma unroll
        for (int i = 0; i < 16; ++i) vf[i] = (_Float16)0.0f;
      }
      oacc[n] = wmma16(af, vf, oacc[n]);
    }
  }

#pragma unroll
  for (int n = 0; n < 4; ++n)
#pragma unroll
    for (int r = 0; r < 8; ++r) {
      int rowg = qt * 16 + hi * 8 + r;
      int colg = h * HD + n * 16 + lo;
      o[((size_t)(bb * Tn + rowg)) * Dm + colg] = (_Float16)(oacc[n][r] / lrow[r]);
    }
}

// ---------------------------------------------------------------------------
// Host-side orchestration
// ---------------------------------------------------------------------------
extern "C" void kernel_launch(void* const* d_in, const int* in_sizes, int n_in,
                              void* d_out, int out_size, void* d_ws, size_t ws_size,
                              hipStream_t stream) {
  const int*   idx    = (const int*)  d_in[0];
  const float* tok    = (const float*)d_in[1];
  const float* pos    = (const float*)d_in[2];
  const float* ln1_g  = (const float*)d_in[3];
  const float* ln1_b  = (const float*)d_in[4];
  const float* wq     = (const float*)d_in[5];
  const float* wk     = (const float*)d_in[6];
  const float* wv     = (const float*)d_in[7];
  const float* proj_w = (const float*)d_in[8];
  const float* proj_b = (const float*)d_in[9];
  const float* ln2_g  = (const float*)d_in[10];
  const float* ln2_b  = (const float*)d_in[11];
  const float* fc1_w  = (const float*)d_in[12];
  const float* fc1_b  = (const float*)d_in[13];
  const float* fc2_w  = (const float*)d_in[14];
  const float* fc2_b  = (const float*)d_in[15];
  const float* lnf_g  = (const float*)d_in[16];
  const float* lnf_b  = (const float*)d_in[17];
  const float* head_w = (const float*)d_in[18];
  const float* head_b = (const float*)d_in[19];

  // ---- workspace carve-up ----
  char* p = (char*)d_ws;
  float* xa = (float*)p;            p += (size_t)Mrows * Dm * 4;
  float* xb = (float*)p;            p += (size_t)Mrows * Dm * 4;
  _Float16* h16  = (_Float16*)p;    p += (size_t)Mrows * Dm * 2;
  _Float16* q16  = (_Float16*)p;    p += (size_t)Mrows * Dm * 2;
  _Float16* k16  = (_Float16*)p;    p += (size_t)Mrows * Dm * 2;
  _Float16* v16  = (_Float16*)p;    p += (size_t)Mrows * Dm * 2;
  _Float16* a16  = (_Float16*)p;    p += (size_t)Mrows * Dm * 2;
  _Float16* m16  = (_Float16*)p;    p += (size_t)Mrows * Fn * 2;
  _Float16* wq16 = (_Float16*)p;    p += (size_t)Ln * Dm * Dm * 2;
  _Float16* wk16 = (_Float16*)p;    p += (size_t)Ln * Dm * Dm * 2;
  _Float16* wv16 = (_Float16*)p;    p += (size_t)Ln * Dm * Dm * 2;
  _Float16* wp16 = (_Float16*)p;    p += (size_t)Ln * Dm * Dm * 2;
  _Float16* f116 = (_Float16*)p;    p += (size_t)Ln * Dm * Fn * 2;
  _Float16* f216 = (_Float16*)p;    p += (size_t)Ln * Dm * Fn * 2;
  _Float16* wh16 = (_Float16*)p;    p += (size_t)Dm * Vn * 2;

  dim3 blk(256);
  const size_t nDD = (size_t)Ln * Dm * Dm;   // 8 Mi
  const size_t nDF = (size_t)Ln * Dm * Fn;   // 32 Mi
  const size_t nHV = (size_t)Dm * Vn;        // 32.77 M

  // ---- one-time (per launch) weight downconversion f32 -> f16 ----
  cvt_kernel<<<(uint32_t)(nDD / 1024), blk, 0, stream>>>(wq,     wq16);
  cvt_kernel<<<(uint32_t)(nDD / 1024), blk, 0, stream>>>(wk,     wk16);
  cvt_kernel<<<(uint32_t)(nDD / 1024), blk, 0, stream>>>(wv,     wv16);
  cvt_kernel<<<(uint32_t)(nDD / 1024), blk, 0, stream>>>(proj_w, wp16);
  cvt_kernel<<<(uint32_t)(nDF / 1024), blk, 0, stream>>>(fc1_w,  f116);
  cvt_kernel<<<(uint32_t)(nDF / 1024), blk, 0, stream>>>(fc2_w,  f216);
  cvt_kernel<<<(uint32_t)(nHV / 1024), blk, 0, stream>>>(head_w, wh16);

  dim3 gD (Dm / 128, Mrows / 128);
  dim3 gF (Fn / 128, Mrows / 128);
  dim3 gV (Vn / 128, Mrows / 128);
  dim3 gAt(Tn / 16, Hn, Bn);

  embed_kernel<<<Mrows, blk, 0, stream>>>(idx, tok, pos, xa);

  for (int l = 0; l < Ln; ++l) {
    const size_t oDD = (size_t)l * Dm * Dm;
    const size_t oDF = (size_t)l * Dm * Fn;
    layernorm_kernel<<<Mrows, blk, 0, stream>>>(xa, ln1_g + l*Dm, ln1_b + l*Dm, h16);
    gemm_kernel<false,false,false,true><<<gD, blk, 0, stream>>>(
        h16, wq16 + oDD, nullptr, nullptr, q16, Dm, Dm);
    gemm_kernel<false,false,false,true><<<gD, blk, 0, stream>>>(
        h16, wk16 + oDD, nullptr, nullptr, k16, Dm, Dm);
    gemm_kernel<false,false,false,true><<<gD, blk, 0, stream>>>(
        h16, wv16 + oDD, nullptr, nullptr, v16, Dm, Dm);
    attn_kernel<<<gAt, dim3(32), 0, stream>>>(q16, k16, v16, a16);
    gemm_kernel<true,false,true,false><<<gD, blk, 0, stream>>>(
        a16, wp16 + oDD, proj_b + l*Dm, xa, xb, Dm, Dm);
    layernorm_kernel<<<Mrows, blk, 0, stream>>>(xb, ln2_g + l*Dm, ln2_b + l*Dm, h16);
    gemm_kernel<true,true,false,true><<<gF, blk, 0, stream>>>(
        h16, f116 + oDF, fc1_b + (size_t)l*Fn, nullptr, m16, Fn, Dm);
    gemm_kernel<true,false,true,false><<<gD, blk, 0, stream>>>(
        m16, f216 + oDF, fc2_b + l*Dm, xb, xa, Dm, Fn);
  }

  layernorm_kernel<<<Mrows, blk, 0, stream>>>(xa, lnf_g, lnf_b, h16);
  gemm_kernel<true,false,false,false><<<gV, blk, 0, stream>>>(
      h16, wh16, head_b, nullptr, (float*)d_out, Vn, Dm);

  (void)in_sizes; (void)n_in; (void)out_size; (void)ws_size;
}
